// BlindSpotConv_39453569581609
// MI455X (gfx1250) — compile-verified, matrix operand order checked by reading, not run on hardware
//
#include <hip/hip_runtime.h>
#include <hip/hip_bf16.h>

typedef __attribute__((ext_vector_type(16))) _Float16 v16h;
typedef __attribute__((ext_vector_type(8)))  _Float16 v8h;
typedef __attribute__((ext_vector_type(8)))  float    v8f;

#define IN_C    64
#define OUT_C   64
#define IMG_H   256
#define IMG_W   256
#define TILE_W  128
#define LDSCOLS 130   // TILE_W + 2 halo columns
#define ICP     72    // ic pitch in halves: 144 B = 36 dwords -> conflict-free b128

// Implicit-GEMM 3x3 conv (pad=1) via v_wmma_f32_16x16x32_f16.
// Workgroup: 128 threads = 4 waves. Each workgroup: one (n, h) row segment of
// 128 pixels x all 64 output channels. Each wave: TWO 16-pixel M-tiles
// (32 pixels) x 64 ocs, so every B fragment fetched from LDS feeds 2 WMMAs.
__global__ __launch_bounds__(128, 1)
void conv3x3_wmma_kernel(const float* __restrict__ x,
                         const float* __restrict__ wgt,
                         const float* __restrict__ bias,
                         float* __restrict__ out)
{
    // f16 staging, ic innermost with pitch 72 halves so each lane's 8-half
    // fragment chunk is one aligned, bank-conflict-free ds_load_b128.
    __shared__ _Float16 lds_in[3 * LDSCOLS * ICP] __attribute__((aligned(16))); // 56,160 B
    __shared__ _Float16 lds_w [9 * OUT_C * ICP]  __attribute__((aligned(16))); // 82,944 B

    const int tid = threadIdx.x;
    const int bid = blockIdx.x;
    const int wt  = bid & 1;            // which 128-pixel half of the row
    const int h   = (bid >> 1) & 255;   // output row
    const int n   = bid >> 9;           // batch
    const int w0  = wt * TILE_W;

    // ---- Stage input halo tile (3 rows x 130 cols x 64 ch), coalesced along w.
    for (int idx = tid; idx < 3 * LDSCOLS * IN_C; idx += 128) {
        const int col = idx % LDSCOLS;
        const int t   = idx / LDSCOLS;
        const int ic  = t % IN_C;
        const int r   = t / IN_C;
        const int hh  = h + r - 1;
        const int ww  = w0 + col - 1;
        float v = 0.0f;
        if (hh >= 0 && hh < IMG_H && ww >= 0 && ww < IMG_W)
            v = x[((n * IN_C + ic) * IMG_H + hh) * IMG_W + ww];
        lds_in[(r * LDSCOLS + col) * ICP + ic] = (_Float16)v;
    }
    // ---- Stage weights OIHW -> [pos][oc][ic] f16; global reads walk the
    // contiguous pos dimension (fully coalesced).
    for (int idx = tid; idx < 9 * OUT_C * IN_C; idx += 128) {
        const int pos = idx % 9;
        const int t   = idx / 9;
        const int ic  = t % IN_C;
        const int oc  = t / IN_C;
        lds_w[(pos * OUT_C + oc) * ICP + ic] = (_Float16)wgt[idx];
    }
    __syncthreads();

    const int wave  = tid >> 5;    // 0..3, owns pixels [wave*32, wave*32+32)
    const int lane  = tid & 31;
    const int laneM = lane & 15;   // A: M row / B,D: N col
    const int hi    = lane >> 4;   // K-half selector per 16-bit A/B VGPR layout

    v8f acc[8] = {};               // [m-tile 0/1][oc-tile 0..3]

    // ISA 16-bit 16x32 fragment layout: lanes 0-15 hold K 0-7 (elems 0-7) and
    // K 16-23 (elems 8-15); lanes 16-31 hold K 8-15 and K 24-31.
#pragma unroll
    for (int kh = 0; kh < 3; ++kh) {
#pragma unroll
        for (int kw = 0; kw < 3; ++kw) {
            const int pos  = kh * 3 + kw;
            const int col0 = wave * 32 + laneM + kw;           // m-tile 0
            const v8h* arow0 = (const v8h*)&lds_in[(kh * LDSCOLS + col0) * ICP];
            const v8h* arow1 = (const v8h*)&lds_in[(kh * LDSCOLS + col0 + 16) * ICP];
#pragma unroll
            for (int c = 0; c < 2; ++c) {                      // ic chunks of 32
                const v16h a0 = __builtin_shufflevector(
                    arow0[c * 4 + hi], arow0[c * 4 + hi + 2],
                    0, 1, 2, 3, 4, 5, 6, 7, 8, 9, 10, 11, 12, 13, 14, 15);
                const v16h a1 = __builtin_shufflevector(
                    arow1[c * 4 + hi], arow1[c * 4 + hi + 2],
                    0, 1, 2, 3, 4, 5, 6, 7, 8, 9, 10, 11, 12, 13, 14, 15);
#pragma unroll
                for (int t = 0; t < 4; ++t) {
                    const v8h* brow =
                        (const v8h*)&lds_w[(pos * OUT_C + (t * 16 + laneM)) * ICP];
                    const v16h b = __builtin_shufflevector(
                        brow[c * 4 + hi], brow[c * 4 + hi + 2],
                        0, 1, 2, 3, 4, 5, 6, 7, 8, 9, 10, 11, 12, 13, 14, 15);
                    acc[t] = __builtin_amdgcn_wmma_f32_16x16x32_f16(
                        false, a0, false, b, (short)0, acc[t], false, false);
                    acc[4 + t] = __builtin_amdgcn_wmma_f32_16x16x32_f16(
                        false, a1, false, b, (short)0, acc[4 + t], false, false);
                }
            }
        }
    }

    // ---- Epilogue: transpose through LDS so global stores are coalesced
    // 128-float channel rows. Reuse the input-tile LDS (33,792 B needed).
    __syncthreads();
    float* lds_out = (float*)lds_in;
    const int PITCH = 132;   // 132 % 64 == 4 -> conflict-free column writes
#pragma unroll
    for (int m = 0; m < 2; ++m) {
#pragma unroll
        for (int t = 0; t < 4; ++t) {
            const int oc = t * 16 + laneM;                 // D: N = lane & 15
#pragma unroll
            for (int r = 0; r < 8; ++r) {                  // D: M = r + 8*hi
                const int pix = wave * 32 + m * 16 + hi * 8 + r;
                lds_out[oc * PITCH + pix] = acc[m * 4 + t][r];
            }
        }
    }
    __syncthreads();
    for (int idx = tid; idx < OUT_C * TILE_W; idx += 128) {
        const int oc  = idx >> 7;
        const int pix = idx & 127;
        out[((n * OUT_C + oc) * IMG_H + h) * IMG_W + w0 + pix] =
            lds_out[oc * PITCH + pix] + bias[oc];
    }
}

extern "C" void kernel_launch(void* const* d_in, const int* in_sizes, int n_in,
                              void* d_out, int out_size, void* d_ws, size_t ws_size,
                              hipStream_t stream) {
    (void)in_sizes; (void)n_in; (void)out_size; (void)d_ws; (void)ws_size;
    const float* x    = (const float*)d_in[0];
    const float* wgt  = (const float*)d_in[1];
    const float* bias = (const float*)d_in[2];
    float* out        = (float*)d_out;

    // 16 batches * 256 rows * 2 half-row tiles = 8192 workgroups
    dim3 grid(16 * 256 * 2);
    dim3 block(128);
    conv3x3_wmma_kernel<<<grid, block, 0, stream>>>(x, wgt, bias, out);
}
